// DetailEncoder_6640019440245
// MI455X (gfx1250) — matrix operand order repaired and checked by reading, not compiled, not run on hardware
//
#include <hip/hip_runtime.h>
#include <hip/hip_bf16.h>

typedef __attribute__((ext_vector_type(16))) __bf16 v16bf;
typedef __attribute__((ext_vector_type(8)))  __bf16 v8bf;
typedef __attribute__((ext_vector_type(8)))  float  v8f;

#define LDS_STRIDE 40   // 16 rows x 32 K halves, padded (80B rows) for bank spread

__device__ __forceinline__ v8f wmma_bf16(v16bf a, v16bf b, v8f c) {
  // emits v_wmma_f32_16x16x32_bf16
  return __builtin_amdgcn_wmma_f32_16x16x32_bf16(false, a, false, b, (short)0, c,
                                                 false, false);
}

__device__ __forceinline__ v16bf combine16(v8bf x0, v8bf x1) {
  v16bf r;
#pragma unroll
  for (int i = 0; i < 8; ++i) { r[i] = x0[i]; r[i + 8] = x1[i]; }
  return r;
}

// A-matrix 16x32 bf16 fragment from LDS (rows padded to LDS_STRIDE halves).
// Layout: lane l holds M=l%16; halves j -> K=16*(l/16)+j (matches B swizzle)
__device__ __forceinline__ v16bf load_afrag(const __bf16* lds, int lane) {
  const int row  = lane & 15;
  const int base = (lane >> 4) << 4;              // 0 or 16
  const v8bf* p0 = (const v8bf*)(lds + row * LDS_STRIDE + base);        // 16B aligned
  const v8bf* p1 = (const v8bf*)(lds + row * LDS_STRIDE + base + 8);    // 16B aligned
  return combine16(*p0, *p1);
}

// B-matrix 32x16 fragment from pre-swizzled global plane: 16 contiguous halves/lane.
__device__ __forceinline__ v16bf load_bfrag(const __bf16* g) {
  const v8bf* p = (const v8bf*)g;
  return combine16(p[0], p[1]);
}

__device__ __forceinline__ float gelu_exact(float x) {
  return 0.5f * x * (1.0f + erff(x * 0.70710678118654752440f));
}

__device__ __forceinline__ void zero8(v8f& a) {
#pragma unroll
  for (int i = 0; i < 8; ++i) a[i] = 0.0f;
}

// ---------------------------------------------------------------------------
// K0: pre-swizzle f32 weight [K x N] (row-major) into bf16 hi/lo planes laid
// out per 32x16 WMMA B tile: dst[((kt*(N/16)+nt)*32 + lane)*16 + j],
// with n = nt*16 + lane%16, k = kt*32 + 16*(lane/16) + j.
// ---------------------------------------------------------------------------
__global__ __launch_bounds__(256) void k_swizzle(const float* __restrict__ W,
                                                 __bf16* __restrict__ hi,
                                                 __bf16* __restrict__ lo,
                                                 int K, int N) {
  const int tid   = blockIdx.x * 256 + threadIdx.x;
  const int NT    = N >> 4;
  const int total = (K >> 5) * NT * 32;
  if (tid >= total) return;
  const int lane = tid & 31;
  const int blk  = tid >> 5;
  const int nt   = blk % NT;
  const int kt   = blk / NT;
  const int n    = (nt << 4) + (lane & 15);
  const int kb   = (kt << 5) + ((lane >> 4) << 4);
  const size_t dbase = ((size_t)blk << 9) + ((size_t)lane << 4);
#pragma unroll
  for (int j = 0; j < 16; ++j) {
    const float f = W[(size_t)(kb + j) * N + n];
    const __bf16 h = (__bf16)f;
    hi[dbase + j] = h;
    lo[dbase + j] = (__bf16)(f - (float)h);
  }
}

// ---------------------------------------------------------------------------
// K1: fused scorer. Per WG: 16 token rows, full Dq=256.
// H1 = H @ Ws1 (bf16 hi/lo split, 3 WMMAs per k-step), then
// score = sum_n gelu(H1+bs1)*Ws2 + bs2, masked.
// ---------------------------------------------------------------------------
__global__ __launch_bounds__(256) void k_score(const float* __restrict__ H,
                                               const int* __restrict__ amask,
                                               const __bf16* __restrict__ whi,
                                               const __bf16* __restrict__ wlo,
                                               const float* __restrict__ bs1,
                                               const float* __restrict__ Ws2,
                                               const float* __restrict__ bs2,
                                               float* __restrict__ scores) {
  __shared__ __bf16 Ahi[16 * LDS_STRIDE];
  __shared__ __bf16 Alo[16 * LDS_STRIDE];
  __shared__ float rowsum[16];
  const int tid = threadIdx.x, lane = tid & 31, wave = tid >> 5;
  const int mbase = blockIdx.x << 4;

  v8f acc[2];
  zero8(acc[0]); zero8(acc[1]);
  if (tid < 16) rowsum[tid] = 0.0f;

  const int arow = tid >> 4;            // 0..15
  const int ak   = (tid & 15) << 1;     // 0..30

  for (int kt = 0; kt < 32; ++kt) {
    const float2 hv = *(const float2*)(H + ((size_t)(mbase + arow) << 10) + (kt << 5) + ak);
    const __bf16 h0 = (__bf16)hv.x, h1 = (__bf16)hv.y;
    Ahi[arow * LDS_STRIDE + ak]     = h0;
    Ahi[arow * LDS_STRIDE + ak + 1] = h1;
    Alo[arow * LDS_STRIDE + ak]     = (__bf16)(hv.x - (float)h0);
    Alo[arow * LDS_STRIDE + ak + 1] = (__bf16)(hv.y - (float)h1);
    __syncthreads();
    const v16bf ahi = load_afrag(Ahi, lane);
    const v16bf alo = load_afrag(Alo, lane);
#pragma unroll
    for (int t = 0; t < 2; ++t) {
      const int ntg = (wave << 1) + t;                              // 0..15
      const size_t boff = (((size_t)(kt * 16 + ntg) << 5) + lane) << 4;
      const v16bf bhi = load_bfrag(whi + boff);
      const v16bf blo = load_bfrag(wlo + boff);
      acc[t] = wmma_bf16(ahi, bhi, acc[t]);
      acc[t] = wmma_bf16(ahi, blo, acc[t]);
      acc[t] = wmma_bf16(alo, bhi, acc[t]);
    }
    __syncthreads();
  }

  const int hl = lane >> 4, nl = lane & 15;
#pragma unroll
  for (int t = 0; t < 2; ++t) {
    const int ng = (((wave << 1) + t) << 4) + nl;                   // 0..255
    const float b1 = bs1[ng];
    const float w2 = Ws2[ng];
#pragma unroll
    for (int r = 0; r < 8; ++r) {
      const float v = gelu_exact(acc[t][r] + b1) * w2;
      atomicAdd(&rowsum[r + (hl << 3)], v);
    }
  }
  __syncthreads();
  if (tid < 16) {
    const int g = mbase + tid;
    float s = rowsum[tid] + bs2[0];
    if (amask[g] == 0) s = -__builtin_inff();
    scores[g] = s;
  }
}

// ---------------------------------------------------------------------------
// K2: top-32 of 4096 per batch, tie-break = lowest index (matches lax.top_k).
// Also writes detail_mask directly.
// ---------------------------------------------------------------------------
__global__ __launch_bounds__(256) void k_topk(const float* __restrict__ scores,
                                              int* __restrict__ topIdx,
                                              float* __restrict__ maskOut) {
  __shared__ float sv[4096];
  __shared__ float rv[256];
  __shared__ int   ri[256];
  const int b = blockIdx.x, tid = threadIdx.x;
  for (int i = tid; i < 4096; i += 256) sv[i] = scores[(b << 12) + i];
  __syncthreads();
  for (int sel = 0; sel < 32; ++sel) {
    const int base = tid << 4;
    float bv = sv[base]; int bi = base;
#pragma unroll
    for (int j = 1; j < 16; ++j) {
      const float v = sv[base + j];
      if (v > bv) { bv = v; bi = base + j; }
    }
    rv[tid] = bv; ri[tid] = bi;
    __syncthreads();
    for (int off = 128; off > 0; off >>= 1) {
      if (tid < off) {
        const float ov = rv[tid + off]; const int oi = ri[tid + off];
        if (ov > rv[tid] || (ov == rv[tid] && oi < ri[tid])) { rv[tid] = ov; ri[tid] = oi; }
      }
      __syncthreads();
    }
    if (tid == 0) {
      topIdx[(b << 5) + sel]  = ri[0];
      maskOut[(b << 5) + sel] = (rv[0] > -__builtin_inff()) ? 1.0f : 0.0f;
      sv[ri[0]] = -__builtin_inff();
    }
    __syncthreads();
  }
}

// ---------------------------------------------------------------------------
// K3: gather + T = gelu(selected @ Wp1 + bp1).  M=256, N=512, K=1024.
// Grid 32: blockIdx = mtile(16) x nhalf(2).
// ---------------------------------------------------------------------------
__global__ __launch_bounds__(256) void k_mlp1(const float* __restrict__ H,
                                              const int* __restrict__ topIdx,
                                              const __bf16* __restrict__ whi,
                                              const __bf16* __restrict__ wlo,
                                              const float* __restrict__ bp1,
                                              float* __restrict__ T) {
  __shared__ __bf16 Ahi[16 * LDS_STRIDE];
  __shared__ __bf16 Alo[16 * LDS_STRIDE];
  __shared__ int rowsrc[16];
  const int tid = threadIdx.x, lane = tid & 31, wave = tid >> 5;
  const int mtile = blockIdx.x & 15, nhalf = blockIdx.x >> 4;
  if (tid < 16) {
    const int m = (mtile << 4) + tid;
    const int b = m >> 5, slot = m & 31;
    rowsrc[tid] = (b << 12) + topIdx[(b << 5) + slot];   // row in flattened [B*S]
  }
  v8f acc[2];
  zero8(acc[0]); zero8(acc[1]);
  const int arow = tid >> 4, ak = (tid & 15) << 1;
  __syncthreads();

  for (int kt = 0; kt < 32; ++kt) {
    const float2 hv = *(const float2*)(H + ((size_t)rowsrc[arow] << 10) + (kt << 5) + ak);
    const __bf16 h0 = (__bf16)hv.x, h1 = (__bf16)hv.y;
    Ahi[arow * LDS_STRIDE + ak]     = h0;
    Ahi[arow * LDS_STRIDE + ak + 1] = h1;
    Alo[arow * LDS_STRIDE + ak]     = (__bf16)(hv.x - (float)h0);
    Alo[arow * LDS_STRIDE + ak + 1] = (__bf16)(hv.y - (float)h1);
    __syncthreads();
    const v16bf ahi = load_afrag(Ahi, lane);
    const v16bf alo = load_afrag(Alo, lane);
#pragma unroll
    for (int t = 0; t < 2; ++t) {
      const int ntg = (nhalf << 4) + (wave << 1) + t;               // 0..31
      const size_t boff = (((size_t)(kt * 32 + ntg) << 5) + lane) << 4;
      const v16bf bhi = load_bfrag(whi + boff);
      const v16bf blo = load_bfrag(wlo + boff);
      acc[t] = wmma_bf16(ahi, bhi, acc[t]);
      acc[t] = wmma_bf16(ahi, blo, acc[t]);
      acc[t] = wmma_bf16(alo, bhi, acc[t]);
    }
    __syncthreads();
  }

  const int hl = lane >> 4, nl = lane & 15;
#pragma unroll
  for (int t = 0; t < 2; ++t) {
    const int ng = (((nhalf << 4) + (wave << 1) + t) << 4) + nl;    // 0..511
    const float b1 = bp1[ng];
#pragma unroll
    for (int r = 0; r < 8; ++r) {
      const int M = r + (hl << 3);
      T[(size_t)((mtile << 4) + M) * 512 + ng] = gelu_exact(acc[t][r] + b1);
    }
  }
}

// ---------------------------------------------------------------------------
// K4: detail = LayerNorm(T @ Wp2 + bp2).  M=256, N=512, K=512. Grid 16.
// ---------------------------------------------------------------------------
__global__ __launch_bounds__(256) void k_mlp2_ln(const float* __restrict__ T,
                                                 const __bf16* __restrict__ whi,
                                                 const __bf16* __restrict__ wlo,
                                                 const float* __restrict__ bp2,
                                                 const float* __restrict__ gamma,
                                                 const float* __restrict__ beta,
                                                 float* __restrict__ out) {
  __shared__ __bf16 Ahi[16 * LDS_STRIDE];
  __shared__ __bf16 Alo[16 * LDS_STRIDE];
  __shared__ float ssum[16], ssq[16], smu[16], srs[16];
  const int tid = threadIdx.x, lane = tid & 31, wave = tid >> 5;
  const int mtile = blockIdx.x;
  v8f acc[4];
#pragma unroll
  for (int t = 0; t < 4; ++t) zero8(acc[t]);
  if (tid < 16) { ssum[tid] = 0.0f; ssq[tid] = 0.0f; }
  const int arow = tid >> 4, ak = (tid & 15) << 1;

  for (int kt = 0; kt < 16; ++kt) {
    const float2 hv = *(const float2*)(T + (size_t)((mtile << 4) + arow) * 512 + (kt << 5) + ak);
    const __bf16 h0 = (__bf16)hv.x, h1 = (__bf16)hv.y;
    Ahi[arow * LDS_STRIDE + ak]     = h0;
    Ahi[arow * LDS_STRIDE + ak + 1] = h1;
    Alo[arow * LDS_STRIDE + ak]     = (__bf16)(hv.x - (float)h0);
    Alo[arow * LDS_STRIDE + ak + 1] = (__bf16)(hv.y - (float)h1);
    __syncthreads();
    const v16bf ahi = load_afrag(Ahi, lane);
    const v16bf alo = load_afrag(Alo, lane);
#pragma unroll
    for (int t = 0; t < 4; ++t) {
      const int ntg = (wave << 2) + t;                              // 0..31
      const size_t boff = (((size_t)(kt * 32 + ntg) << 5) + lane) << 4;
      const v16bf bhi = load_bfrag(whi + boff);
      const v16bf blo = load_bfrag(wlo + boff);
      acc[t] = wmma_bf16(ahi, bhi, acc[t]);
      acc[t] = wmma_bf16(ahi, blo, acc[t]);
      acc[t] = wmma_bf16(alo, bhi, acc[t]);
    }
    __syncthreads();
  }

  const int hl = lane >> 4, nl = lane & 15;
#pragma unroll
  for (int t = 0; t < 4; ++t) {
    const int ng = (((wave << 2) + t) << 4) + nl;
    const float b2 = bp2[ng];
#pragma unroll
    for (int r = 0; r < 8; ++r) {
      acc[t][r] += b2;
      const float v = acc[t][r];
      atomicAdd(&ssum[r + (hl << 3)], v);
      atomicAdd(&ssq[r + (hl << 3)], v * v);
    }
  }
  __syncthreads();
  if (tid < 16) {
    const float mu = ssum[tid] * (1.0f / 512.0f);
    float var = ssq[tid] * (1.0f / 512.0f) - mu * mu;
    if (var < 0.0f) var = 0.0f;
    smu[tid] = mu;
    srs[tid] = 1.0f / sqrtf(var + 1e-5f);
  }
  __syncthreads();
#pragma unroll
  for (int t = 0; t < 4; ++t) {
    const int ng = (((wave << 2) + t) << 4) + nl;
    const float g = gamma[ng], bb = beta[ng];
#pragma unroll
    for (int r = 0; r < 8; ++r) {
      const int M = r + (hl << 3);
      out[(size_t)((mtile << 4) + M) * 512 + ng] = (acc[t][r] - smu[M]) * srs[M] * g + bb;
    }
  }
}

// ---------------------------------------------------------------------------
extern "C" void kernel_launch(void* const* d_in, const int* in_sizes, int n_in,
                              void* d_out, int out_size, void* d_ws, size_t ws_size,
                              hipStream_t stream) {
  (void)in_sizes; (void)n_in; (void)out_size; (void)ws_size;
  const float* H     = (const float*)d_in[0];
  const int*   am    = (const int*)d_in[1];
  const float* Ws1   = (const float*)d_in[2];
  const float* bs1   = (const float*)d_in[3];
  const float* Ws2   = (const float*)d_in[4];
  const float* bs2   = (const float*)d_in[5];
  const float* Wp1   = (const float*)d_in[6];
  const float* bp1   = (const float*)d_in[7];
  const float* Wp2   = (const float*)d_in[8];
  const float* bp2   = (const float*)d_in[9];
  const float* gamma = (const float*)d_in[10];
  const float* beta  = (const float*)d_in[11];
  float* out = (float*)d_out;

  char* ws = (char*)d_ws;
  __bf16* w1hi = (__bf16*)(ws);                       // 1024*256 halves
  __bf16* w1lo = w1hi + 1024 * 256;
  __bf16* p1hi = (__bf16*)(ws + (1u << 20));          // 1024*512 halves
  __bf16* p1lo = p1hi + 1024 * 512;
  __bf16* p2hi = (__bf16*)(ws + 3u * (1u << 20));     // 512*512 halves
  __bf16* p2lo = p2hi + 512 * 512;
  float*  scores = (float*)(ws + 4u * (1u << 20));    // 32768 f32
  int*    topIdx = (int*)(ws + 4u * (1u << 20) + 524288u);
  float*  T      = (float*)(ws + 5u * (1u << 20));    // 256*512 f32

  k_swizzle<<<64,  256, 0, stream>>>(Ws1, w1hi, w1lo, 1024, 256);
  k_swizzle<<<128, 256, 0, stream>>>(Wp1, p1hi, p1lo, 1024, 512);
  k_swizzle<<<64,  256, 0, stream>>>(Wp2, p2hi, p2lo, 512, 512);
  k_score<<<2048, 256, 0, stream>>>(H, am, w1hi, w1lo, bs1, Ws2, bs2, scores);
  k_topk<<<8, 256, 0, stream>>>(scores, topIdx, out + 8 * 32 * 512);
  k_mlp1<<<32, 256, 0, stream>>>(H, topIdx, p1hi, p1lo, bp1, T);
  k_mlp2_ln<<<16, 256, 0, stream>>>(T, p2hi, p2lo, bp2, gamma, beta, out);
}
